// PathwayAutoencoderWithTissue_47931835023414
// MI455X (gfx1250) — matrix operand order repaired and checked by reading, not compiled, not run on hardware
//
#include <hip/hip_runtime.h>
#include <hip/hip_bf16.h>
#include <math.h>

// ---------------------------------------------------------------------------
// PathwayAutoencoderWithTissue on MI455X (gfx1250, wave32, WMMA)
//
// Pipeline:
//   K1: gather x[:,pathway_idx] -> f16, per-pathway GEMM via v_wmma_f32_16x16x32_f16,
//       bias+ReLU, H -> ws (f16, coalesced b128 stores via LDS re-read),
//       attention scores -> ws (f32)
//   K2: softmax over 200 pathways, context, 128->32->2 MLP, l2norm, angle
//   K3: rec = l2norm(pcn @ Wdec + bdec) streamed over G=20000
// ---------------------------------------------------------------------------

#define N_SAMP   2048      // B*NS = 16*128
#define G_DIM    20000
#define P_DIM    200
#define S_DIM    128
#define E_DIM    128
#define NT       16        // samples per tile (WMMA M)
#define LDA      136       // padded LDS stride (halves) for A  (136*2B = 272B, breaks bank pattern)
#define LDB      136       // padded LDS stride (halves) for B
#define LDH      132       // padded LDS stride (floats) for H / Q
#define TWO_PI_F 6.283185307179586f

typedef _Float16 v8h  __attribute__((ext_vector_type(8)));
typedef _Float16 v16h __attribute__((ext_vector_type(16)));
typedef float    v8f  __attribute__((ext_vector_type(8)));

// ---------------------------------------------------------------------------
// Kernel 1: fused gather + pathway projection (WMMA) + ReLU + attention scores
// grid = (N_SAMP/NT, P_DIM), block = 256 (8 waves)
// Each wave computes one 16(samples) x 16(E) output tile, K = 128 in 4 wmma steps.
// ---------------------------------------------------------------------------
__global__ __launch_bounds__(256)
void k1_proj_scores(const float* __restrict__ x,            // (N, G)
                    const int*   __restrict__ tissue_idx,   // (N,)
                    const int*   __restrict__ pathway_idx,  // (P, S)
                    const float* __restrict__ W_proj,       // (P, E, S)
                    const float* __restrict__ b_proj,       // (P, E)
                    const float* __restrict__ tissue_table, // (T, E)
                    _Float16*    __restrict__ H16,          // (N, P, E) f16  [ws]
                    float*       __restrict__ scores)       // (N, P)         [ws]
{
  __shared__ int      sIdx[S_DIM];
  __shared__ _Float16 sA[NT * LDA];        // gathered tile, (m, s) row-major
  __shared__ _Float16 sB[S_DIM * LDB];     // W_p as (e, s), s contiguous
  __shared__ float    sQ[NT * LDH];        // tissue embeddings for tile
  __shared__ float    sH[NT * LDH];        // ReLU'd projection tile (f32)
  __shared__ float    sBias[E_DIM];

  const int t  = threadIdx.x;
  const int p  = blockIdx.y;
  const int n0 = blockIdx.x * NT;

  if (t < S_DIM) {
    sIdx[t]  = pathway_idx[p * S_DIM + t];
    sBias[t] = b_proj[p * E_DIM + t];
  }

  // W_proj[p] (E,S) row-major -> sB[(e, s)] as f16 (coalesced global read)
  const float* Wp = W_proj + (size_t)p * E_DIM * S_DIM;
  for (int i = t; i < E_DIM * S_DIM; i += 256) {
    const int e = i >> 7;        // 0..127
    const int s = i & 127;
    sB[e * LDB + s] = (_Float16)Wp[i];
  }

  // Q rows for this sample tile
  for (int i = t; i < NT * E_DIM; i += 256) {
    const int m = i >> 7;
    const int e = i & 127;
    const int tix = tissue_idx[n0 + m];
    sQ[m * LDH + e] = tissue_table[(size_t)tix * E_DIM + e];
  }
  __syncthreads();   // sIdx ready before gather; sB ready before WMMA

  // gather: sA[m][s] = x[n0+m, pathway_idx[p,s]]  (L2-resident: x fits in 192MB L2)
  for (int i = t; i < NT * S_DIM; i += 256) {
    const int m = i >> 7;
    const int s = i & 127;
    sA[m * LDA + s] = (_Float16)x[(size_t)(n0 + m) * G_DIM + sIdx[s]];
  }
  __syncthreads();

  // ---- WMMA: D(16x16 f32) += A(16x32 f16) * B(32x16 f16), 4 K-steps ----
  const int lane = t & 31;
  const int w    = t >> 5;            // wave id 0..7 -> E slice
  const int e0   = w * 16;
  const int am   = lane & 15;         // A row (sample)
  const int hi   = lane >> 4;         // lane half (ISA A-layout K split)

  v8f c = {};
#pragma unroll
  for (int kk = 0; kk < 4; ++kk) {
    const int k0 = kk * 32;
    // A fragment (16x32): lane<16 -> K {k0..k0+7, k0+16..k0+23}; lane>=16 -> +8
    const v8h a_lo = *(const v8h*)&sA[am * LDA + k0 + hi * 8];
    const v8h a_hi = *(const v8h*)&sA[am * LDA + k0 + 16 + hi * 8];
    const v16h a = __builtin_shufflevector(a_lo, a_hi,
        0,1,2,3,4,5,6,7,8,9,10,11,12,13,14,15);
    // B fragment (32x16): N = e0+(lane&15), K same split as A
    const v8h b_lo = *(const v8h*)&sB[(e0 + am) * LDB + k0 + hi * 8];
    const v8h b_hi = *(const v8h*)&sB[(e0 + am) * LDB + k0 + 16 + hi * 8];
    const v16h b = __builtin_shufflevector(b_lo, b_hi,
        0,1,2,3,4,5,6,7,8,9,10,11,12,13,14,15);
    c = __builtin_amdgcn_wmma_f32_16x16x32_f16(
        /*neg_a=*/false, a, /*neg_b=*/false, b,
        /*c_mod=*/(short)0, c, /*reuse_a=*/false, /*reuse_b=*/false);
  }

  // C/D layout: N = lane&15, M = r + 8*hi  -> bias + ReLU into LDS only
  const int  e    = e0 + (lane & 15);
  const float bias = sBias[e];
#pragma unroll
  for (int r = 0; r < 8; ++r) {
    const int mr = r + hi * 8;
    float hv = c[r] + bias;
    hv = hv > 0.f ? hv : 0.f;
    sH[mr * LDH + e] = hv;
  }
  __syncthreads();

  // Re-read sH row-major: coalesced f16 spill of H (one b128 store per thread,
  // 256B contiguous per sample row) + attention scores.
  const int sm = t >> 4;      // sample 0..15
  const int li = t & 15;

  v8h hrow;
#pragma unroll
  for (int j = 0; j < 8; ++j)
    hrow[j] = (_Float16)sH[sm * LDH + li * 8 + j];
  *(v8h*)&H16[((size_t)(n0 + sm) * P_DIM + p) * E_DIM + li * 8] = hrow;

  // scores[n,p] = (Q[n] . H[n,p]) / sqrt(E) ; 16 lanes per sample, 8 e's each
  float partial = 0.f;
#pragma unroll
  for (int j = 0; j < 8; ++j) {
    const int e2 = li + j * 16;
    partial += sH[sm * LDH + e2] * sQ[sm * LDH + e2];
  }
  partial += __shfl_xor(partial, 1, 32);
  partial += __shfl_xor(partial, 2, 32);
  partial += __shfl_xor(partial, 4, 32);
  partial += __shfl_xor(partial, 8, 32);
  if (li == 0)
    scores[(size_t)(n0 + sm) * P_DIM + p] = partial * 0.08838834764831845f; // 1/sqrt(128)
}

// ---------------------------------------------------------------------------
// Kernel 2: softmax over pathways, context, MLP head, l2norm + angle
// grid = N_SAMP blocks, 256 threads
// ---------------------------------------------------------------------------
__global__ __launch_bounds__(256)
void k2_attn_head(const float*    __restrict__ scores,  // (N, P)
                  const _Float16* __restrict__ H16,     // (N, P, E)
                  const float* __restrict__ W1,         // (E, 32)
                  const float* __restrict__ b1,         // (32,)
                  const float* __restrict__ W2,         // (32, 2)
                  const float* __restrict__ b2,         // (2,)
                  float* __restrict__ out_pcn,          // (N, 2)
                  float* __restrict__ out_ang)          // (N,)
{
  __shared__ float sS[P_DIM];
  __shared__ float sCtxPart[2][E_DIM];
  __shared__ float sCtx[E_DIM];
  __shared__ float sRed[8];
  __shared__ float sHd[32];
  __shared__ float sStat;

  const int t = threadIdx.x;
  const int n = blockIdx.x;

  if (t < P_DIM) sS[t] = scores[(size_t)n * P_DIM + t];
  __syncthreads();

  // max over 200 scores
  float v = (t < P_DIM) ? sS[t] : -1e30f;
  for (int off = 16; off; off >>= 1) v = fmaxf(v, __shfl_xor(v, off, 32));
  if ((t & 31) == 0) sRed[t >> 5] = v;
  __syncthreads();
  if (t == 0) {
    float mx = sRed[0];
    for (int i = 1; i < 8; ++i) mx = fmaxf(mx, sRed[i]);
    sStat = mx;
  }
  __syncthreads();
  const float mx = sStat;

  // exp + sum
  float ex = 0.f;
  if (t < P_DIM) { ex = __expf(sS[t] - mx); sS[t] = ex; }
  float sv = ex;
  for (int off = 16; off; off >>= 1) sv += __shfl_xor(sv, off, 32);
  if ((t & 31) == 0) sRed[t >> 5] = sv;
  __syncthreads();
  if (t == 0) {
    float s = 0.f;
    for (int i = 0; i < 8; ++i) s += sRed[i];
    sStat = s;
  }
  __syncthreads();
  const float invsum = 1.0f / sStat;

  // context[e] = sum_p attn[p] * H[n,p,e]   (two half-ranges of p per e)
  const int e    = t & 127;
  const int half = t >> 7;
  const _Float16* Hrow = H16 + ((size_t)n * P_DIM + half * 100) * E_DIM + e;
  float acc = 0.f;
  for (int p = 0; p < 100; ++p)
    acc += sS[half * 100 + p] * (float)Hrow[(size_t)p * E_DIM];
  sCtxPart[half][e] = acc;
  __syncthreads();
  if (t < E_DIM) sCtx[t] = (sCtxPart[0][t] + sCtxPart[1][t]) * invsum;
  __syncthreads();

  // h = relu(ctx @ W1 + b1)
  if (t < 32) {
    float h = b1[t];
    for (int e2 = 0; e2 < E_DIM; ++e2) h += sCtx[e2] * W1[e2 * 32 + t];
    sHd[t] = h > 0.f ? h : 0.f;
  }
  __syncthreads();

  // pc = h @ W2 + b2
  if (t < 2) {
    float pc = b2[t];
    for (int j = 0; j < 32; ++j) pc += sHd[j] * W2[j * 2 + t];
    sRed[t] = pc;
  }
  __syncthreads();

  if (t == 0) {
    float p0 = sRed[0], p1 = sRed[1];
    const float nrm = sqrtf(p0 * p0 + p1 * p1);
    const float inv = 1.0f / fmaxf(nrm, 1e-12f);
    p0 *= inv; p1 *= inv;
    out_pcn[2 * n]     = p0;
    out_pcn[2 * n + 1] = p1;
    float ang = atan2f(p1, p0) + TWO_PI_F;
    ang = fmodf(ang, TWO_PI_F);   // arg > 0 here, fmod == remainder
    out_ang[n] = ang;
  }
}

// ---------------------------------------------------------------------------
// Kernel 3: rec = l2norm(pcn @ Wdec + bdec) over G=20000.
// Two streaming passes; Wdec/bdec (240KB) stay resident in L2/WGP$,
// so HBM cost ~= the 164MB rec write.
// grid = N_SAMP blocks, 256 threads
// ---------------------------------------------------------------------------
__global__ __launch_bounds__(256)
void k3_decode(const float* __restrict__ pcn,   // (N, 2) — read back from d_out
               const float* __restrict__ Wdec,  // (2, G)
               const float* __restrict__ bdec,  // (G,)
               float* __restrict__ rec)         // (N, G)
{
  __shared__ float sRed[8];
  __shared__ float sInv;
  const int t = threadIdx.x;
  const int n = blockIdx.x;
  const float p0 = pcn[2 * n];
  const float p1 = pcn[2 * n + 1];

  float ss = 0.f;
  for (int g = t; g < G_DIM; g += 256) {
    const float v = fmaf(p0, Wdec[g], fmaf(p1, Wdec[G_DIM + g], bdec[g]));
    ss += v * v;
  }
  for (int off = 16; off; off >>= 1) ss += __shfl_xor(ss, off, 32);
  if ((t & 31) == 0) sRed[t >> 5] = ss;
  __syncthreads();
  if (t == 0) {
    float s = 0.f;
    for (int i = 0; i < 8; ++i) s += sRed[i];
    sInv = 1.0f / fmaxf(sqrtf(s), 1e-12f);
  }
  __syncthreads();
  const float inv = sInv;
  float* out = rec + (size_t)n * G_DIM;
  for (int g = t; g < G_DIM; g += 256) {
    const float v = fmaf(p0, Wdec[g], fmaf(p1, Wdec[G_DIM + g], bdec[g]));
    out[g] = v * inv;
  }
}

// ---------------------------------------------------------------------------
extern "C" void kernel_launch(void* const* d_in, const int* in_sizes, int n_in,
                              void* d_out, int out_size, void* d_ws, size_t ws_size,
                              hipStream_t stream) {
  (void)in_sizes; (void)n_in; (void)out_size; (void)ws_size;

  const float* x            = (const float*)d_in[0];   // (16,128,20000)
  const int*   tissue_idx   = (const int*)  d_in[1];   // (16,128)
  const int*   pathway_idx  = (const int*)  d_in[2];   // (200,128)
  const float* W_proj       = (const float*)d_in[3];   // (200,128,128)
  const float* b_proj       = (const float*)d_in[4];   // (200,128)
  const float* tissue_table = (const float*)d_in[5];   // (50,128)
  const float* W1           = (const float*)d_in[6];   // (128,32)
  const float* b1           = (const float*)d_in[7];   // (32,)
  const float* W2           = (const float*)d_in[8];   // (32,2)
  const float* b2           = (const float*)d_in[9];   // (2,)
  const float* Wdec         = (const float*)d_in[10];  // (2,20000)
  const float* bdec         = (const float*)d_in[11];  // (20000,)

  float* out     = (float*)d_out;
  float* out_pcn = out;                       // N*2
  float* out_ang = out + (size_t)N_SAMP * 2;  // N
  float* out_rec = out + (size_t)N_SAMP * 3;  // N*G

  // workspace: scores (N*P f32) then H (N*P*E f16)  -> ~107 MB total
  float*    scores = (float*)d_ws;
  _Float16* H16    = (_Float16*)((char*)d_ws + (size_t)N_SAMP * P_DIM * sizeof(float));

  dim3 g1(N_SAMP / NT, P_DIM);   // (128, 200)
  k1_proj_scores<<<g1, 256, 0, stream>>>(x, tissue_idx, pathway_idx, W_proj, b_proj,
                                         tissue_table, H16, scores);
  k2_attn_head<<<N_SAMP, 256, 0, stream>>>(scores, H16, W1, b1, W2, b2,
                                           out_pcn, out_ang);
  k3_decode<<<N_SAMP, 256, 0, stream>>>(out_pcn, Wdec, bdec, out_rec);
}
